// MCA_90520730730559
// MI455X (gfx1250) — matrix operand order repaired and checked by reading, not compiled, not run on hardware
//
#include <hip/hip_runtime.h>
#include <stdint.h>

// ---------------------------------------------------------------------------
// Problem constants (match reference)
// ---------------------------------------------------------------------------
#define T1N   2048
#define T2N   2048
#define BATCH 4
#define EMB   1024
#define MROWS (T1N * BATCH)   // 8192, row index m = t*BATCH + b

// GEMM tiling
#define BM 64                 // block tile M
#define BN 256                // block tile N
#define BK 32                 // k-step per stage
#define LSTR 40               // padded LDS row stride (elements) -> conflict-free
#define A_ELEMS (BM * LSTR)   // 2560
#define STAGE   (A_ELEMS + BN * LSTR)   // 12800 elements per stage
#define STAGE_BYTES (STAGE * 2)         // 25600 bytes

typedef __attribute__((ext_vector_type(16))) __bf16 v16bf;
typedef __attribute__((ext_vector_type(8)))  __bf16 bf16x8;
typedef __attribute__((ext_vector_type(4)))  __bf16 bf16x4;
typedef __attribute__((ext_vector_type(8)))  float  v8f;
typedef __attribute__((ext_vector_type(4)))  float  f32x4;

// round-to-nearest-even f32 -> bf16
static __device__ __forceinline__ __bf16 f2bf(float f) {
    unsigned u = __builtin_bit_cast(unsigned, f);
    unsigned r = (u + 0x7FFFu + ((u >> 16) & 1u)) >> 16;
    unsigned short h = (unsigned short)r;
    return __builtin_bit_cast(__bf16, h);
}

static __device__ __forceinline__ void stv(float* p, float v)  { *p = v; }
static __device__ __forceinline__ void stv(__bf16* p, float v) { *p = f2bf(v); }

// ---------------------------------------------------------------------------
// CDNA5 async copy engine: 16B global -> LDS per lane, tracked by ASYNCcnt.
// lofs = LDS byte address (low 32 bits of generic LDS pointer, ISA 10.2).
// ---------------------------------------------------------------------------
static __device__ __forceinline__ void async_copy16(unsigned lofs, const __bf16* g) {
    asm volatile("global_load_async_to_lds_b128 %0, %1, off"
                 :: "v"(lofs), "v"((unsigned long long)(uintptr_t)g)
                 : "memory");
}
static __device__ __forceinline__ void wait_async0() {
    asm volatile("s_wait_asynccnt 0" ::: "memory");
}

// ---------------------------------------------------------------------------
// f32 -> bf16 convert, 4-wide (b128 load, b64 store); n4 = count/4
// ---------------------------------------------------------------------------
__global__ __launch_bounds__(256) void cvt_f32_bf16_v4(const f32x4* __restrict__ in,
                                                       bf16x4* __restrict__ out,
                                                       int64_t n4) {
    int64_t i = (int64_t)blockIdx.x * blockDim.x + threadIdx.x;
    int64_t stride = (int64_t)gridDim.x * blockDim.x;
    for (; i < n4; i += stride) {
        f32x4 x = in[i];
        bf16x4 y;
#pragma unroll
        for (int e = 0; e < 4; ++e) y[e] = f2bf(x[e]);
        out[i] = y;
    }
}

// ---------------------------------------------------------------------------
// LDS double-buffered strided BF16 GEMM:  D[z] = scale * (A[z] @ W[z]^T) + bias
//   A: [M,K] bf16, row stride lda, batch stride sAz
//   W: [N,K] bf16, row stride ldw, batch stride sWz
//   D: [M,N] TOUT,  row stride ldd, batch stride sDz
// 256 threads = 8 waves; block tile 64M x 256N x 32K (double buffered, 50KB LDS);
// wave tile 32M x 64N (2 A-frags x 4 B-frags -> 8 WMMAs per stage).
// Global->LDS via async copy engine (ASYNCcnt); per-thread DMA addresses are
// pointer-incremented (constant 64B/stage), LDS destination ping-pongs via a
// single XOR toggle. Fragments via ds_load_b128. Requires 64|M, 256|N, 32|K.
//
// Fragment layouts per CDNA5 ISA 7.12.2 (wave32), half = lane>>4:
//   A 16x32 bf16: row M=lane&15; elems 0..7 -> K=half*8+e ; 8..15 -> K=16+half*8+e
//   B 32x16 bf16: col N=lane&15; elems 0..7 -> K=half*8+e ; 8..15 -> K=16+half*8+e
// ---------------------------------------------------------------------------
template <typename TOUT>
__global__ __launch_bounds__(256) void gemm_bf16_wmma(
    const __bf16* __restrict__ A, int64_t lda, int64_t sAz,
    const __bf16* __restrict__ W, int64_t ldw, int64_t sWz,
    TOUT* __restrict__ D, int64_t ldd, int64_t sDz,
    const float* __restrict__ bias, float scale,
    int M, int N, int K)
{
    __shared__ __bf16 smem[2 * STAGE];   // 51200 bytes

    const int tid  = threadIdx.x;
    const int lane = tid & 31;
    const int wave = tid >> 5;
    const int half = lane >> 4;
    const int l15  = lane & 15;
    const int wm   = wave & 1;           // 2 waves in M
    const int wn   = wave >> 1;          // 4 waves in N
    const int z    = blockIdx.z;

    const __bf16* Ablk = A + (int64_t)z * sAz + (int64_t)blockIdx.y * BM * lda;
    const __bf16* Wblk = W + (int64_t)z * sWz + (int64_t)blockIdx.x * BN * ldw;

    // ---- per-thread DMA addressing (computed once) -----------------------
    // 16B chunks, 4 per 32-element row: A = 1 chunk/thread, B = 4 chunks/thread
    const int arow = tid >> 2;
    const int ako  = (tid & 3) * 8;
    const __bf16* gA = Ablk + (int64_t)arow * lda + ako;          // += BK/stage
    unsigned lA = (unsigned)(uintptr_t)(smem + arow * LSTR + ako);
    const __bf16* gW[4];
    unsigned lW[4];
#pragma unroll
    for (int j = 0; j < 4; ++j) {
        const int c = tid + j * 256;
        const int row = c >> 2, ko = (c & 3) * 8;
        gW[j] = Wblk + (int64_t)row * ldw + ko;
        lW[j] = (unsigned)(uintptr_t)(smem + A_ELEMS + row * LSTR + ko);
    }

    // ---- compute-side LDS fragment bases (stage 0) -----------------------
    const __bf16* fA = smem + (wm * 32 + l15) * LSTR + half * 8;
    const __bf16* fB = smem + A_ELEMS + (wn * 64 + l15) * LSTR + half * 8;

    v8f acc[2][4];
#pragma unroll
    for (int mt = 0; mt < 2; ++mt)
#pragma unroll
        for (int nt = 0; nt < 4; ++nt) acc[mt][nt] = (v8f){};

    const int KT = K / BK;

    // ---- preload stage 0 -------------------------------------------------
    async_copy16(lA, gA); gA += BK;
#pragma unroll
    for (int j = 0; j < 4; ++j) { async_copy16(lW[j], gW[j]); gW[j] += BK; }
    wait_async0();
    __syncthreads();

    // ---- main loop: DMA stage kt+1 while computing stage kt --------------
    unsigned sel = STAGE_BYTES;          // LDS dst offset for next stage
    for (int kt = 0; kt < KT; ++kt) {
        if (kt + 1 < KT) {
            async_copy16(lA + sel, gA); gA += BK;
#pragma unroll
            for (int j = 0; j < 4; ++j) { async_copy16(lW[j] + sel, gW[j]); gW[j] += BK; }
            sel ^= STAGE_BYTES;
        }

        const int so = (kt & 1) * STAGE;   // element offset of current stage
        v16bf a[2], b[4];
#pragma unroll
        for (int mt = 0; mt < 2; ++mt) {
            const __bf16* ar = fA + so + mt * (16 * LSTR);
            bf16x8 lo = *(const bf16x8*)ar;
            bf16x8 hi = *(const bf16x8*)(ar + 16);
            a[mt] = __builtin_shufflevector(lo, hi,
                        0,1,2,3,4,5,6,7,8,9,10,11,12,13,14,15);
        }
#pragma unroll
        for (int nt = 0; nt < 4; ++nt) {
            const __bf16* br = fB + so + nt * (16 * LSTR);
            bf16x8 lo = *(const bf16x8*)br;
            bf16x8 hi = *(const bf16x8*)(br + 16);
            b[nt] = __builtin_shufflevector(lo, hi,
                        0,1,2,3,4,5,6,7,8,9,10,11,12,13,14,15);
        }
#pragma unroll
        for (int mt = 0; mt < 2; ++mt)
#pragma unroll
            for (int nt = 0; nt < 4; ++nt)
                acc[mt][nt] = __builtin_amdgcn_wmma_f32_16x16x32_bf16(
                    false, a[mt], false, b[nt], (short)0, acc[mt][nt], false, false);

        wait_async0();      // next-stage DMA done (overlapped with WMMAs above)
        __syncthreads();
    }

    // ---- epilogue: C/D layout -> M = r + 8*half, N = lane&15 -------------
    TOUT* Dz = D + (int64_t)z * sDz;
    const int m0 = blockIdx.y * BM + wm * 32;
    const int n0 = blockIdx.x * BN + wn * 64;
#pragma unroll
    for (int mt = 0; mt < 2; ++mt) {
#pragma unroll
        for (int nt = 0; nt < 4; ++nt) {
            const int n  = n0 + nt * 16 + l15;
            const float bv = bias ? bias[n] : 0.0f;
#pragma unroll
            for (int r = 0; r < 8; ++r) {
                const int m = m0 + mt * 16 + half * 8 + r;
                stv(Dz + (int64_t)m * ldd + n, acc[mt][nt][r] * scale + bv);
            }
        }
    }
}

// ---------------------------------------------------------------------------
// Row softmax over T2 f32 scores; writes bf16 probabilities IN PLACE into the
// first half of the same f32 row (so P has row stride 2*T2 bf16 elements).
// One block (256 thr) per (b,t) row; each thread owns 8 contiguous elements
// (two b128 loads, one 16B bf16 store).
// ---------------------------------------------------------------------------
__global__ __launch_bounds__(256) void softmax_row_bf16(float* __restrict__ S) {
    float* row = S + (int64_t)blockIdx.x * T2N;
    const int tid  = threadIdx.x;
    const int wave = tid >> 5;
    __shared__ float sred[8];

    const f32x4* r4 = (const f32x4*)(row + tid * 8);
    f32x4 u0 = r4[0], u1 = r4[1];
    float v[8];
#pragma unroll
    for (int i = 0; i < 4; ++i) { v[i] = u0[i]; v[i + 4] = u1[i]; }

    float mx = v[0];
#pragma unroll
    for (int i = 1; i < 8; ++i) mx = fmaxf(mx, v[i]);
#pragma unroll
    for (int off = 16; off > 0; off >>= 1) mx = fmaxf(mx, __shfl_xor(mx, off, 32));
    if ((tid & 31) == 0) sred[wave] = mx;
    __syncthreads();
    float m2 = sred[0];
#pragma unroll
    for (int i = 1; i < 8; ++i) m2 = fmaxf(m2, sred[i]);

    float s = 0.0f;
#pragma unroll
    for (int i = 0; i < 8; ++i) { v[i] = __expf(v[i] - m2); s += v[i]; }
#pragma unroll
    for (int off = 16; off > 0; off >>= 1) s += __shfl_xor(s, off, 32);
    __syncthreads();                 // protect sred before reuse
    if ((tid & 31) == 0) sred[wave] = s;
    __syncthreads();
    float tot = 0.0f;
#pragma unroll
    for (int i = 0; i < 8; ++i) tot += sred[i];
    const float inv = 1.0f / tot;

    // all block reads of `row` completed before the barriers above
    bf16x8 pv;
#pragma unroll
    for (int i = 0; i < 8; ++i) pv[i] = f2bf(v[i] * inv);
    *(bf16x8*)((__bf16*)row + tid * 8) = pv;
}

// ---------------------------------------------------------------------------
// V transpose: V[(s*BATCH+b)*EMB + c]  ->  Vt[b][c][s]  (bf16, LDS-tiled)
// grid (T2/32, EMB/32, BATCH), 256 threads = 32x8
// ---------------------------------------------------------------------------
__global__ __launch_bounds__(256) void transpose_V(const __bf16* __restrict__ V,
                                                   __bf16* __restrict__ Vt) {
    __shared__ __bf16 tile[32][33];
    const int b  = blockIdx.z;
    const int s0 = blockIdx.x * 32;
    const int c0 = blockIdx.y * 32;
    const int tx = threadIdx.x & 31;
    const int ty = threadIdx.x >> 5;
#pragma unroll
    for (int j = 0; j < 4; ++j) {
        const int s = s0 + ty + j * 8;
        tile[ty + j * 8][tx] = V[((int64_t)s * BATCH + b) * EMB + c0 + tx];
    }
    __syncthreads();
#pragma unroll
    for (int j = 0; j < 4; ++j) {
        const int c = c0 + ty + j * 8;
        Vt[(int64_t)b * EMB * T2N + (int64_t)c * T2N + s0 + tx] = tile[tx][ty + j * 8];
    }
}

// ---------------------------------------------------------------------------
// Host-side orchestration
// ---------------------------------------------------------------------------
extern "C" void kernel_launch(void* const* d_in, const int* in_sizes, int n_in,
                              void* d_out, int out_size, void* d_ws, size_t ws_size,
                              hipStream_t stream) {
    const float* x1 = (const float*)d_in[0];
    const float* x2 = (const float*)d_in[1];
    // d_in[2] = x3 (unused by reference forward)
    const float* wq = (const float*)d_in[3];
    const float* bq = (const float*)d_in[4];
    const float* wk = (const float*)d_in[5];
    const float* bk = (const float*)d_in[6];
    const float* wv = (const float*)d_in[7];
    const float* bv = (const float*)d_in[8];
    const float* wu = (const float*)d_in[9];
    const float* bu = (const float*)d_in[10];
    float* out = (float*)d_out;

    // workspace carve-up (bytes)
    char* ws = (char*)d_ws;
    const int64_t nX = (int64_t)MROWS * EMB;   // 8,388,608 elements
    const int64_t nW = (int64_t)EMB * EMB;     // 1,048,576 elements
    __bf16* X1b = (__bf16*)ws; ws += nX * 2;
    __bf16* X2b = (__bf16*)ws; ws += nX * 2;
    __bf16* Wqb = (__bf16*)ws; ws += nW * 2;
    __bf16* Wkb = (__bf16*)ws; ws += nW * 2;
    __bf16* Wvb = (__bf16*)ws; ws += nW * 2;
    __bf16* Wub = (__bf16*)ws; ws += nW * 2;
    __bf16* Qb  = (__bf16*)ws; ws += nX * 2;
    __bf16* Kb  = (__bf16*)ws; ws += nX * 2;
    __bf16* Vb  = (__bf16*)ws; ws += nX * 2;
    __bf16* Vt  = (__bf16*)ws; ws += nX * 2;
    __bf16* Ob  = (__bf16*)ws; ws += nX * 2;
    float*  S   = (float*)ws;  ws += (int64_t)BATCH * T1N * T2N * 4;  // 64 MB

    // 1) convert inputs + weights to bf16 (4-wide)
    cvt_f32_bf16_v4<<<1024, 256, 0, stream>>>((const f32x4*)x1, (bf16x4*)X1b, nX / 4);
    cvt_f32_bf16_v4<<<1024, 256, 0, stream>>>((const f32x4*)x2, (bf16x4*)X2b, nX / 4);
    cvt_f32_bf16_v4<<<256, 256, 0, stream>>>((const f32x4*)wq, (bf16x4*)Wqb, nW / 4);
    cvt_f32_bf16_v4<<<256, 256, 0, stream>>>((const f32x4*)wk, (bf16x4*)Wkb, nW / 4);
    cvt_f32_bf16_v4<<<256, 256, 0, stream>>>((const f32x4*)wv, (bf16x4*)Wvb, nW / 4);
    cvt_f32_bf16_v4<<<256, 256, 0, stream>>>((const f32x4*)wu, (bf16x4*)Wub, nW / 4);

    // 2) projections: Q/K/V = X @ W^T + b   (M=8192, N=1024, K=1024)
    {
        dim3 grid(EMB / BN, MROWS / BM, 1);
        gemm_bf16_wmma<__bf16><<<grid, 256, 0, stream>>>(
            X1b, EMB, 0, Wqb, EMB, 0, Qb, EMB, 0, bq, 1.0f, MROWS, EMB, EMB);
        gemm_bf16_wmma<__bf16><<<grid, 256, 0, stream>>>(
            X2b, EMB, 0, Wkb, EMB, 0, Kb, EMB, 0, bk, 1.0f, MROWS, EMB, EMB);
        gemm_bf16_wmma<__bf16><<<grid, 256, 0, stream>>>(
            X2b, EMB, 0, Wvb, EMB, 0, Vb, EMB, 0, bv, 1.0f, MROWS, EMB, EMB);
    }

    // 3) V transpose for contiguous B-operand loads in attn@V
    {
        dim3 grid(T2N / 32, EMB / 32, BATCH);
        transpose_V<<<grid, 256, 0, stream>>>(Vb, Vt);
    }

    // 4) scores: S[b] = (Q_b @ K_b^T) / sqrt(C)   (batched, M=N=2048, K=1024)
    {
        dim3 grid(T2N / BN, T1N / BM, BATCH);
        gemm_bf16_wmma<float><<<grid, 256, 0, stream>>>(
            Qb, (int64_t)BATCH * EMB, EMB,          // A = Q rows of batch b (stride B*C)
            Kb, (int64_t)BATCH * EMB, EMB,          // W = K rows of batch b
            S, T2N, (int64_t)T1N * T2N,
            nullptr, 0.03125f /* 1/sqrt(1024) */, T1N, T2N, EMB);
    }

    // 5) softmax rows; P (bf16) written in place over S rows (row stride 2*T2)
    softmax_row_bf16<<<BATCH * T1N, 256, 0, stream>>>(S);

    // 6) attn @ V: O[b] = P_b @ V_b   (M=2048, N=1024, K=2048)
    {
        dim3 grid(EMB / BN, T1N / BM, BATCH);
        gemm_bf16_wmma<__bf16><<<grid, 256, 0, stream>>>(
            (const __bf16*)S, (int64_t)2 * T2N, (int64_t)T1N * 2 * T2N, // P
            Vt, T2N, (int64_t)EMB * T2N,                                // Vt[b]: [C, T2]
            Ob, (int64_t)BATCH * EMB, EMB,       // O row t of batch b at (t*B+b)*C
            nullptr, 1.0f, T1N, EMB, T2N);
    }

    // 7) final projection: out = O @ Wu^T + bu, f32 straight to d_out
    //    (O laid out as contiguous [8192,1024] with row index t*B+b, which is
    //     exactly the [T1,B,C] output layout)
    {
        dim3 grid(EMB / BN, MROWS / BM, 1);
        gemm_bf16_wmma<float><<<grid, 256, 0, stream>>>(
            Ob, EMB, 0, Wub, EMB, 0, out, EMB, 0, bu, 1.0f, MROWS, EMB, EMB);
    }
}
